// SemanticGNN_sent_45973329936661
// MI455X (gfx1250) — matrix operand order repaired and checked by reading, not compiled, not run on hardware
//
#include <hip/hip_runtime.h>
#include <math.h>

// ---------------------------------------------------------------------------
// SemanticGNN forward for MI455X (gfx1250, wave32, WMMA 16x16x32 f16->f32).
// Stages: scan -> pad(+f16,+inv-norm) -> attn/w (TDM-staged LDS) -> Gv -> Y
//         -> Z -> BN stats -> BN+residual max reduce.
// ~75 GFLOP total; intermediates sized to live in the 192MB L2.
// ---------------------------------------------------------------------------

typedef __attribute__((ext_vector_type(16))) _Float16 v16h;
typedef __attribute__((ext_vector_type(8)))  _Float16 v8h;
typedef __attribute__((ext_vector_type(8)))  float    v8f;
typedef unsigned int u32x4 __attribute__((ext_vector_type(4)));
typedef int          i32x4 __attribute__((ext_vector_type(4)));
typedef int          i32x8 __attribute__((ext_vector_type(8)));

#define D_DIM 1024
#define INTER 512
#define L64   64

#if __has_builtin(__builtin_amdgcn_tensor_load_to_lds)
#define HAVE_TDM 1
#elif __has_builtin(__builtin_amdgcn_global_load_async_to_lds_b128)
#define HAVE_ASYNC 1
#endif

// ---- WMMA helpers ----------------------------------------------------------

static __device__ __forceinline__ v8f wmma16(v16h a, v16h b, v8f c) {
  // D = A(16x32 f16) * B(32x16 f16) + C(16x16 f32)
  return __builtin_amdgcn_wmma_f32_16x16x32_f16(
      /*neg_a=*/false, a, /*neg_b=*/false, b,
      /*c_mod=*/(short)0, c, /*reuse_a=*/false, /*reuse_b=*/false);
}

// A-matrix 16x32 f16 tile from a row-major matrix (lda in elements).
// lanes 0-15: row=lane,    K = 0..15
// lanes16-31: row=lane-16, K = 16..31
static __device__ __forceinline__ v16h load_a_tile(const _Float16* __restrict__ src,
                                                   int lda) {
  const int lane = threadIdx.x & 31;
  const int row  = lane & 15;
  const int koff = (lane >> 4) << 4;  // 0 or 16
  const _Float16* p = src + (size_t)row * lda + koff;
  v8h lo = *reinterpret_cast<const v8h*>(p);
  v8h hi = *reinterpret_cast<const v8h*>(p + 8);
  v16h a;
#pragma unroll
  for (int i = 0; i < 8; ++i) { a[i] = lo[i]; a[i + 8] = hi[i]; }
  return a;
}

// B-matrix 32x16 f16 tile. Source layout: row n of `src` holds column n of B
// (i.e. src is the transpose of the mathematical B, contiguous in K).
// lanes 0-15: col=lane, K=0..15 ; lanes 16-31: col=lane-16, K=16..31
static __device__ __forceinline__ v16h load_b_tile(const _Float16* __restrict__ src,
                                                   int ldb) {
  const int lane = threadIdx.x & 31;
  const int col  = lane & 15;
  const int koff = (lane >> 4) << 4;  // 0 or 16
  return *reinterpret_cast<const v16h*>(src + (size_t)col * ldb + koff);
}

// ---- Tensor Data Mover: DMA one [64 x 1024] f16 row-major block to LDS ----
// D# per CDNA5 ISA: group0 = flags/lds/global/type,
// group1 = data_size + tensor/tile dims + dim0 stride. 2D tensor -> groups
// 2 & 3 disabled (zero). Issued by one wave; TENSORcnt waited by that wave.

#if defined(HAVE_TDM)
static __device__ __forceinline__ void tdm_load_block(const _Float16* gsrc,
                                                      _Float16* lds_dst) {
  const unsigned long long ga = (unsigned long long)(uintptr_t)gsrc;
  const unsigned int       la = (unsigned int)(uintptr_t)lds_dst;  // LDS offset
  u32x4 g0;
  g0[0] = 1u;                                            // count=1 (valid D#)
  g0[1] = la;                                            // lds_addr
  g0[2] = (unsigned int)(ga & 0xFFFFFFFFu);              // global_addr[31:0]
  g0[3] = (unsigned int)((ga >> 32) & 0x01FFFFFFu)       // global_addr[56:32]
        | (2u << 30);                                    // type=2 ("image")
  i32x8 g1;
  g1[0] = (int)(1u << 16);          // data_size=1 (2 bytes), mask=0
  g1[1] = (int)(1024u << 16);       // tensor_dim0[15:0] = 1024 (elements)
  g1[2] = (int)(64u << 16);         // tensor_dim0[31:16]=0 | tensor_dim1=64
  g1[3] = (int)(1024u << 16);       // tensor_dim1[31:16]=0 | tile_dim0=1024
  g1[4] = 64;                       // tile_dim1=64, tile_dim2=0
  g1[5] = 1024;                     // tensor_dim0_stride[31:0] = 1024
  g1[6] = 0;                        // stride hi / dim1_stride
  g1[7] = 0;
  i32x4 z4 = {0, 0, 0, 0};
#if defined(__clang_major__) && (__clang_major__ >= 23)
  i32x8 z8 = {0, 0, 0, 0, 0, 0, 0, 0};
  __builtin_amdgcn_tensor_load_to_lds(g0, g1, z4, z4, z8, 0);
#else
  __builtin_amdgcn_tensor_load_to_lds(g0, g1, z4, z4, 0);
#endif
#if __has_builtin(__builtin_amdgcn_s_wait_tensorcnt)
  __builtin_amdgcn_s_wait_tensorcnt(0);
#else
  asm volatile("s_wait_tensorcnt 0x0" ::: "memory");
#endif
}
#endif

// ---- Stage 0: exclusive scan of sentence counts ---------------------------

__global__ void k_scan(const int* __restrict__ sem, int* __restrict__ starts, int B) {
  if (blockIdx.x == 0 && threadIdx.x == 0) {
    int acc = 0;
    for (int b = 0; b < B; ++b) { starts[b] = acc; acc += sem[b]; }
  }
}

// ---- Stage 1: weight conversion f32 -> f16 --------------------------------

__global__ void k_conv(const float* __restrict__ a, const float* __restrict__ b,
                       _Float16* __restrict__ ah, _Float16* __restrict__ bh, int n) {
  int i = blockIdx.x * blockDim.x + threadIdx.x;
  if (i < n) { ah[i] = (_Float16)a[i]; bh[i] = (_Float16)b[i]; }
}

// ---- Stage 2: pad sentences, convert to f16, 1/||x||^2 --------------------

__global__ void __launch_bounds__(256)
k_pad(const float* __restrict__ sent, const float* __restrict__ glb,
      const int* __restrict__ sem, const int* __restrict__ starts,
      _Float16* __restrict__ xh, float* __restrict__ inv_n2) {
  const int b = blockIdx.x >> 6;
  const int l = blockIdx.x & 63;
  const int n = sem[b];
  const float* src = nullptr;
  if (l < n)                 src = sent + (size_t)(starts[b] + l) * D_DIM;
  else if (l == 0 && n == 0) src = glb + (size_t)b * D_DIM;  // empty -> global feat
  _Float16* dst = xh + ((size_t)b * L64 + l) * D_DIM;
  float ss = 0.f;
  for (int d = threadIdx.x; d < D_DIM; d += blockDim.x) {
    float v = src ? src[d] : 0.f;
    dst[d] = (_Float16)v;
    ss += v * v;
  }
  __shared__ float red[256];
  red[threadIdx.x] = ss;
  __syncthreads();
  for (int s = 128; s > 0; s >>= 1) {
    if (threadIdx.x < s) red[threadIdx.x] += red[threadIdx.x + s];
    __syncthreads();
  }
  // n_clipped^2 = max(norm^2, 1e-16)  (reference: clip(norm,1e-8) then /n/n)
  if (threadIdx.x == 0) inv_n2[b * L64 + l] = 1.f / fmaxf(red[0], 1e-16f);
}

// ---- Stage 3: attn = X*X^T ; w = attn * inv_n2[j] * cap_mask --------------
// grid(B), 256 threads (8 waves). X[b] (64x1024 f16 = 128KB) is DMA'd into
// LDS once per workgroup by the TDM, then both WMMA operands stream from LDS.
// Wave -> (m-tile = wid&3, 2 n-tiles), K=1024.

__global__ void __launch_bounds__(256)
k_attn(const _Float16* __restrict__ xh, const float* __restrict__ inv_n2,
       const float* __restrict__ cap_mask, _Float16* __restrict__ wmat, int Lr) {
  extern __shared__ _Float16 xs[];   // 64*1024 f16, dynamic LDS (128KB)
  const int b    = blockIdx.x;
  const int wid  = threadIdx.x >> 5;
  const int lane = threadIdx.x & 31;
  const _Float16* xb = xh + (size_t)b * L64 * D_DIM;

#if defined(HAVE_TDM)
  if (wid == 0) tdm_load_block(xb, xs);            // async DMA + tensorcnt wait
#elif defined(HAVE_ASYNC)
  {
    typedef __attribute__((address_space(1))) const char gch;
    typedef __attribute__((address_space(3))) char lch;
    gch* gp = (gch*)(uintptr_t)xb;
    lch* lp = (lch*)(uintptr_t)(unsigned int)(uintptr_t)xs;
    for (int i = threadIdx.x; i < (L64 * D_DIM) / 8; i += blockDim.x)
      __builtin_amdgcn_global_load_async_to_lds_b128(gp + (size_t)i * 16,
                                                     lp + (size_t)i * 16, 0, 0);
    asm volatile("s_wait_asynccnt 0x0" ::: "memory");
  }
#else
  {
    const uint4* s4 = reinterpret_cast<const uint4*>(xb);
    uint4* d4 = reinterpret_cast<uint4*>(xs);
    for (int i = threadIdx.x; i < (L64 * D_DIM) / 8; i += blockDim.x) d4[i] = s4[i];
  }
#endif
  __syncthreads();

  const int mt  = wid & 3;
  const int nt0 = (wid >> 2) << 1;                 // 2 N-tiles per wave
  const _Float16* arow = xs + (size_t)(mt * 16) * D_DIM;
  v8f acc[2] = {};
  for (int kb = 0; kb < D_DIM / 32; ++kb) {
    v16h a = load_a_tile(arow + kb * 32, D_DIM);
#pragma unroll
    for (int t = 0; t < 2; ++t) {
      v16h bt = load_b_tile(xs + (size_t)((nt0 + t) * 16) * D_DIM + kb * 32, D_DIM);
      acc[t] = wmma16(a, bt, acc[t]);
    }
  }

  const int nlane = lane & 15;
  const int mbase = (lane >> 4) << 3;
  const float* mb = cap_mask + (size_t)b * Lr * Lr;
  _Float16* wb = wmat + (size_t)b * L64 * L64;
#pragma unroll
  for (int t = 0; t < 2; ++t) {
    const int j   = (nt0 + t) * 16 + nlane;
    const float s = inv_n2[b * L64 + j];
#pragma unroll
    for (int r = 0; r < 8; ++r) {
      const int i = mt * 16 + mbase + r;
      const float mv = (i < Lr && j < Lr) ? mb[(size_t)i * Lr + j] : 0.f;
      wb[(size_t)i * L64 + j] = (_Float16)(acc[t][r] * s * mv);
    }
  }
}

// ---- Stage 4: Gv^T[c][l] = (X * g_w^T + g_b) transposed -------------------
// grid(B,4), 256 threads (8 waves): wave -> (m-tile = wid&3, 4 c-tiles).

__global__ void __launch_bounds__(256)
k_gv(const _Float16* __restrict__ xh, const _Float16* __restrict__ gwh,
     const float* __restrict__ gb, _Float16* __restrict__ gvT) {
  const int b    = blockIdx.x;
  const int wid  = threadIdx.x >> 5;
  const int lane = threadIdx.x & 31;
  const int mt   = wid & 3;
  const int ct0  = blockIdx.y * 8 + (wid >> 2) * 4;
  const _Float16* xb = xh + (size_t)b * L64 * D_DIM + (size_t)(mt * 16) * D_DIM;
  v8f acc[4] = {};
  for (int kb = 0; kb < D_DIM / 32; ++kb) {
    __builtin_prefetch(xb + (kb + 1) * 32, 0, 0);   // global_prefetch_b8
    v16h a = load_a_tile(xb + kb * 32, D_DIM);
#pragma unroll
    for (int t = 0; t < 4; ++t) {
      v16h bt = load_b_tile(gwh + (size_t)((ct0 + t) * 16) * D_DIM + kb * 32, D_DIM);
      acc[t] = wmma16(a, bt, acc[t]);
    }
  }
  const int nlane = lane & 15;
  const int mbase = (lane >> 4) << 3;
  _Float16* gvb = gvT + (size_t)b * INTER * L64;
#pragma unroll
  for (int t = 0; t < 4; ++t) {
    const int c     = (ct0 + t) * 16 + nlane;
    const float bia = gb[c];
#pragma unroll
    for (int r = 0; r < 8; ++r)
      gvb[(size_t)c * L64 + mt * 16 + mbase + r] = (_Float16)(acc[t][r] + bia);
  }
}

// ---- Stage 5: Y[i][c] = w * Gv  (K=64) ------------------------------------

__global__ void __launch_bounds__(256)
k_y(const _Float16* __restrict__ wmat, const _Float16* __restrict__ gvT,
    _Float16* __restrict__ Y) {
  const int b    = blockIdx.x;
  const int wid  = threadIdx.x >> 5;
  const int lane = threadIdx.x & 31;
  const int mt   = wid & 3;
  const int ct0  = blockIdx.y * 8 + (wid >> 2) * 4;
  const _Float16* wb  = wmat + (size_t)b * L64 * L64 + (size_t)(mt * 16) * L64;
  const _Float16* gvb = gvT + (size_t)b * INTER * L64;
  v8f acc[4] = {};
#pragma unroll
  for (int kb = 0; kb < L64 / 32; ++kb) {
    v16h a = load_a_tile(wb + kb * 32, L64);
#pragma unroll
    for (int t = 0; t < 4; ++t) {
      v16h bt = load_b_tile(gvb + (size_t)((ct0 + t) * 16) * L64 + kb * 32, L64);
      acc[t] = wmma16(a, bt, acc[t]);
    }
  }
  const int nlane = lane & 15;
  const int mbase = (lane >> 4) << 3;
  _Float16* yb = Y + (size_t)b * L64 * INTER;
#pragma unroll
  for (int t = 0; t < 4; ++t) {
    const int c = (ct0 + t) * 16 + nlane;
#pragma unroll
    for (int r = 0; r < 8; ++r)
      yb[(size_t)(mt * 16 + mbase + r) * INTER + c] = (_Float16)acc[t][r];
  }
}

// ---- Stage 6: Z[d][i] = Y * W_w^T + W_b  (K=512) --------------------------

__global__ void __launch_bounds__(256)
k_z(const _Float16* __restrict__ Y, const _Float16* __restrict__ wwh,
    const float* __restrict__ wbias, float* __restrict__ Z) {
  const int b    = blockIdx.x;
  const int wid  = threadIdx.x >> 5;
  const int lane = threadIdx.x & 31;
  const int mt   = wid & 3;
  const int dt0  = blockIdx.y * 8 + (wid >> 2) * 4;
  const _Float16* yb = Y + (size_t)b * L64 * INTER + (size_t)(mt * 16) * INTER;
  v8f acc[4] = {};
  for (int kb = 0; kb < INTER / 32; ++kb) {
    __builtin_prefetch(yb + (kb + 1) * 32, 0, 0);
    v16h a = load_a_tile(yb + kb * 32, INTER);
#pragma unroll
    for (int t = 0; t < 4; ++t) {
      v16h bt = load_b_tile(wwh + (size_t)((dt0 + t) * 16) * INTER + kb * 32, INTER);
      acc[t] = wmma16(a, bt, acc[t]);
    }
  }
  const int nlane = lane & 15;
  const int mbase = (lane >> 4) << 3;
  float* zb = Z + (size_t)b * D_DIM * L64;
#pragma unroll
  for (int t = 0; t < 4; ++t) {
    const int d     = (dt0 + t) * 16 + nlane;
    const float bia = wbias[d];
#pragma unroll
    for (int r = 0; r < 8; ++r)
      zb[(size_t)d * L64 + mt * 16 + mbase + r] = acc[t][r] + bia;
  }
}

// ---- Stage 7: BatchNorm batch stats over (B, L) per channel ---------------

__global__ void __launch_bounds__(256)
k_bn(const float* __restrict__ Z, int B, int Lr,
     float* __restrict__ meanv, float* __restrict__ istdv) {
  const int d = blockIdx.x;
  float s = 0.f, ss = 0.f;
  for (int b = threadIdx.x; b < B; b += blockDim.x) {
    const float* p = Z + ((size_t)b * D_DIM + d) * L64;
    for (int i = 0; i < Lr; ++i) { float v = p[i]; s += v; ss += v * v; }
  }
  __shared__ float rs[256], rq[256];
  rs[threadIdx.x] = s; rq[threadIdx.x] = ss;
  __syncthreads();
  for (int st = 128; st > 0; st >>= 1) {
    if (threadIdx.x < st) {
      rs[threadIdx.x] += rs[threadIdx.x + st];
      rq[threadIdx.x] += rq[threadIdx.x + st];
    }
    __syncthreads();
  }
  if (threadIdx.x == 0) {
    const float cnt = (float)B * (float)Lr;
    const float m = rs[0] / cnt;
    const float var = rq[0] / cnt - m * m;   // population variance
    meanv[d] = m;
    istdv[d] = rsqrtf(var + 1e-5f);
  }
}

// ---- Stage 8: out[b][d] = max_i ( BN(Z)[b][d][i] + x[b][i][d] ) -----------

__global__ void __launch_bounds__(256)
k_out(const float* __restrict__ Z, const _Float16* __restrict__ xh,
      const float* __restrict__ meanv, const float* __restrict__ istdv,
      const float* __restrict__ gamma, const float* __restrict__ beta,
      int Lr, float* __restrict__ out) {
  const int b = blockIdx.x;
  const int d = blockIdx.y * 256 + threadIdx.x;
  const float m = meanv[d], is = istdv[d], g = gamma[d], be = beta[d];
  const float* zp = Z + ((size_t)b * D_DIM + d) * L64;
  const _Float16* xp = xh + (size_t)b * L64 * D_DIM + d;
  float best = -INFINITY;
  for (int i = 0; i < Lr; ++i) {
    const float bn = g * (zp[i] - m) * is + be;
    best = fmaxf(best, bn + (float)xp[(size_t)i * D_DIM]);
  }
  out[(size_t)b * D_DIM + d] = best;
}

// ---------------------------------------------------------------------------

extern "C" void kernel_launch(void* const* d_in, const int* in_sizes, int n_in,
                              void* d_out, int out_size, void* d_ws, size_t ws_size,
                              hipStream_t stream) {
  (void)n_in; (void)out_size; (void)ws_size;
  const float* sent  = (const float*)d_in[0];
  const float* glb   = (const float*)d_in[1];
  const float* mask  = (const float*)d_in[2];
  const float* g_w   = (const float*)d_in[3];
  const float* g_b   = (const float*)d_in[4];
  const float* W_w   = (const float*)d_in[5];
  const float* W_b   = (const float*)d_in[6];
  const float* gamma = (const float*)d_in[7];
  const float* beta  = (const float*)d_in[8];
  const int*   sem   = (const int*)d_in[9];
  float* out = (float*)d_out;

  const int B = in_sizes[9];                           // 512
  int Lr = (int)(sqrt((double)in_sizes[2] / (double)B) + 0.5);
  if (Lr > L64) Lr = L64;

  // workspace carve-out (every region fully written before it is read)
  char* p = (char*)d_ws;
  auto carve = [&](size_t bytes) -> char* {
    char* r = p; p += (bytes + 255) & ~(size_t)255; return r;
  };
  int*      starts = (int*)      carve((size_t)B * 4);
  _Float16* gwh    = (_Float16*) carve((size_t)INTER * D_DIM * 2);
  _Float16* wwh    = (_Float16*) carve((size_t)D_DIM * INTER * 2);
  _Float16* xh     = (_Float16*) carve((size_t)B * L64 * D_DIM * 2);
  float*    inv_n2 = (float*)    carve((size_t)B * L64 * 4);
  _Float16* wmat   = (_Float16*) carve((size_t)B * L64 * L64 * 2);
  _Float16* gvT    = (_Float16*) carve((size_t)B * INTER * L64 * 2);
  _Float16* Ybuf   = (_Float16*) carve((size_t)B * L64 * INTER * 2);
  float*    Zbuf   = (float*)    carve((size_t)B * D_DIM * L64 * 4);
  float*    meanv  = (float*)    carve((size_t)D_DIM * 4);
  float*    istdv  = (float*)    carve((size_t)D_DIM * 4);

  k_scan<<<1, 32, 0, stream>>>(sem, starts, B);
  k_conv<<<(INTER * D_DIM + 255) / 256, 256, 0, stream>>>(g_w, W_w, gwh, wwh,
                                                          INTER * D_DIM);
  k_pad<<<B * L64, 256, 0, stream>>>(sent, glb, sem, starts, xh, inv_n2);
  k_attn<<<B, 256, (size_t)L64 * D_DIM * sizeof(_Float16), stream>>>(
      xh, inv_n2, mask, wmat, Lr);
  k_gv<<<dim3(B, 4), 256, 0, stream>>>(xh, gwh, g_b, gvT);
  k_y<<<dim3(B, 4), 256, 0, stream>>>(wmat, gvT, Ybuf);
  k_z<<<dim3(B, 8), 256, 0, stream>>>(Ybuf, wwh, W_b, Zbuf);
  k_bn<<<D_DIM, 256, 0, stream>>>(Zbuf, B, Lr, meanv, istdv);
  k_out<<<dim3(B, D_DIM / 256), 256, 0, stream>>>(Zbuf, xh, meanv, istdv,
                                                  gamma, beta, Lr, out);
}
